// DNRIModel_67164698575426
// MI455X (gfx1250) — matrix operand order repaired and checked by reading, not compiled, not run on hardware
//
#include <hip/hip_runtime.h>
#include <hip/hip_bf16.h>
#include <math.h>

// dims from reference
#define NN   100
#define HH   64
#define TT   2
#define INSZ 8
#define BB   32
#define EE   (NN * (NN - 1))      // 9900 edges
#define TILES   ((EE + 15) / 16)  // 619 16-edge tiles
#define BLKS_PB ((TILES + 7) / 8) // 78 blocks per batch (8 waves/block)

typedef _Float16 v16h __attribute__((ext_vector_type(16)));
typedef _Float16 v8h  __attribute__((ext_vector_type(8)));
typedef float    v8f  __attribute__((ext_vector_type(8)));

union U16h { v16h v; v8h h[2]; };

// ---------------------------------------------------------------- RNG (deterministic)
__device__ __forceinline__ float splitmix_u01(unsigned int x) {
  x += 0x9e3779b9u;
  x ^= x >> 16; x *= 0x21f0aaadu;
  x ^= x >> 15; x *= 0x735a2d97u;
  x ^= x >> 15;
  return ((float)(x >> 8) + 0.5f) * (1.0f / 16777216.0f); // (0,1)
}

// ---------------------------------------------------------------- K0: hard gumbel gate
// T=2, skip_first_edge_type => only edges[:,:,1] matters and it is exactly 0/1.
__global__ __launch_bounds__(256) void gate_kernel(const float* __restrict__ logits,
                                                   float* __restrict__ gate) {
  int idx = blockIdx.x * blockDim.x + threadIdx.x;
  if (idx >= BB * EE) return;
  float l0 = logits[2 * idx + 0];
  float l1 = logits[2 * idx + 1];
  float g0 = -logf(-logf(splitmix_u01(2u * (unsigned)idx + 0u)));
  float g1 = -logf(-logf(splitmix_u01(2u * (unsigned)idx + 1u)));
  // argmax of softmax((l+g)/tau) == argmax of (l+g); ties -> index 0
  gate[idx] = ((l1 + g1) > (l0 + g0)) ? 1.0f : 0.0f;
}

// ---------------------------------------------------------------- K1: hidden -> f16
__global__ __launch_bounds__(256) void h2f16_kernel(const float* __restrict__ h,
                                                    _Float16* __restrict__ h16, int n) {
  int i = blockIdx.x * blockDim.x + threadIdx.x;
  if (i < n) h16[i] = (_Float16)h[i];
}

// ---------------------------------------------------------------- K2: edge message MLP (WMMA)
// Per wave: 16 edges x 64 outputs. GEMM1: K=128 (recv||send), GEMM2: K=64.
// A layout (16-bit 16x32): lane l holds row M=l&15; halves j: K=(j<8?j:j+8)+8*(l>>4).
// B layout (16-bit 32x16): lane l holds col N=l&15; halves j: K=(j<8?j:j+8)+8*(l>>4) (same as A).
// C/D f32 16x16: lane l holds col N=l&15; vgpr v holds row M=v+8*(l>>4).
__global__ __launch_bounds__(256) void edge_mlp_kernel(
    const _Float16* __restrict__ h16, const float* __restrict__ gate,
    const float* __restrict__ fc1w, const float* __restrict__ fc1b,
    const float* __restrict__ fc2w, const float* __restrict__ fc2b,
    const int* __restrict__ send_e, const int* __restrict__ recv_e,
    _Float16* __restrict__ msgs) {
  __shared__ _Float16 sW1[8192];      // fc1_w[0,1] fragments: [kk(4)][nt(4)][lane(32)][16]
  __shared__ _Float16 sW2[4096];      // fc2_w[0,1] fragments: [kk(2)][nt(4)][lane(32)][16]
  __shared__ _Float16 sM1[8 * 1024];  // per-wave 16x64 f16 m1 tile

  const int tid  = threadIdx.x;
  const int b    = blockIdx.x / BLKS_PB;
  const int tblk = blockIdx.x % BLKS_PB;

  // Stage swizzled weight B-fragments into LDS (once per block; hot in L2 after block 0).
  for (int idx = tid; idx < 8192; idx += 256) {
    int j = idx & 15, ln = (idx >> 4) & 31, nt = (idx >> 9) & 3, kk = idx >> 11;
    int K = kk * 32 + (ln >> 4) * 8 + (j & 7) + (j >> 3) * 16;
    sW1[idx] = (_Float16)fc1w[8192 + K * 64 + nt * 16 + (ln & 15)]; // fc1_w[0,1,K,n]
  }
  for (int idx = tid; idx < 4096; idx += 256) {
    int j = idx & 15, ln = (idx >> 4) & 31, nt = (idx >> 9) & 3, kk = idx >> 11;
    int K = kk * 32 + (ln >> 4) * 8 + (j & 7) + (j >> 3) * 16;
    sW2[idx] = (_Float16)fc2w[4096 + K * 64 + nt * 16 + (ln & 15)]; // fc2_w[0,1,K,n]
  }
  __syncthreads();

  const int wave = tid >> 5, lane = tid & 31;
  const int t = tblk * 8 + wave;
  if (t * 16 >= EE) return;           // wave-uniform: EXEC stays all-1 for WMMA
  const int e0  = t * 16;
  const int row = lane & 15, hi = lane >> 4, col0 = lane & 15;

  int eA = e0 + row; if (eA >= EE) eA = EE - 1;   // clamp gather rows of tail tile
  const int nr = recv_e[eA], ns = send_e[eA];
  const _Float16* hr = h16 + ((size_t)b * NN + nr) * HH;
  const _Float16* hs = h16 + ((size_t)b * NN + ns) * HH;

  // GEMM1: [16 x 128] @ [128 x 64]
  v8f acc[4] = {};
#pragma unroll
  for (int kk = 0; kk < 4; ++kk) {
    const _Float16* hp = ((kk < 2) ? hr : hs) + (kk & 1) * 32 + hi * 8;
    U16h A; A.h[0] = *(const v8h*)hp; A.h[1] = *(const v8h*)(hp + 16);
#pragma unroll
    for (int nt = 0; nt < 4; ++nt) {
      const _Float16* wp = &sW1[((kk * 4 + nt) * 32 + lane) * 16];
      U16h Bf; Bf.h[0] = *(const v8h*)wp; Bf.h[1] = *(const v8h*)(wp + 8);
      acc[nt] = __builtin_amdgcn_wmma_f32_16x16x32_f16(
          false, A.v, false, Bf.v, (short)0, acc[nt], false, false);
    }
  }

  // bias + tanh -> per-wave LDS tile (D-layout -> row-major 16x64)
  _Float16* m1 = &sM1[wave * 1024];
#pragma unroll
  for (int nt = 0; nt < 4; ++nt) {
    float bv = fc1b[64 + nt * 16 + col0];
#pragma unroll
    for (int v = 0; v < 8; ++v)
      m1[(v + 8 * hi) * 64 + nt * 16 + col0] = (_Float16)tanhf(acc[nt][v] + bv);
  }
  asm volatile("s_wait_dscnt 0" ::: "memory"); // DS RAW fence within the wave

  // GEMM2: [16 x 64] @ [64 x 64]
  v8f acc2[4] = {};
#pragma unroll
  for (int kk = 0; kk < 2; ++kk) {
    const _Float16* ap = m1 + row * 64 + kk * 32 + hi * 8;
    U16h A; A.h[0] = *(const v8h*)ap; A.h[1] = *(const v8h*)(ap + 16);
#pragma unroll
    for (int nt = 0; nt < 4; ++nt) {
      const _Float16* wp = &sW2[((kk * 4 + nt) * 32 + lane) * 16];
      U16h Bf; Bf.h[0] = *(const v8h*)wp; Bf.h[1] = *(const v8h*)(wp + 8);
      acc2[nt] = __builtin_amdgcn_wmma_f32_16x16x32_f16(
          false, A.v, false, Bf.v, (short)0, acc2[nt], false, false);
    }
  }

  // bias + tanh, gate, store gated message (norm = T-1 = 1)
#pragma unroll
  for (int nt = 0; nt < 4; ++nt) {
    const int ncol = nt * 16 + col0;
    const float bv = fc2b[64 + ncol];
#pragma unroll
    for (int v = 0; v < 8; ++v) {
      const int e  = e0 + v + 8 * hi;
      const int ec = (e < EE) ? e : (EE - 1);
      const float g   = gate[(size_t)b * EE + ec];
      const float val = tanhf(acc2[nt][v] + bv) * g;
      if (e < EE) msgs[((size_t)b * EE + e) * HH + ncol] = (_Float16)val;
    }
  }
}

// ---------------------------------------------------------------- K3: scatter-mean (gather form)
// Full graph: recv-degree == 99; edge(s -> n) = s*99 + n - (n > s).
__global__ __launch_bounds__(256) void agg_kernel(const _Float16* __restrict__ msgs,
                                                  float* __restrict__ agg) {
  int idx = blockIdx.x * blockDim.x + threadIdx.x;
  if (idx >= BB * NN * HH) return;
  int h = idx & 63;
  int n = (idx >> 6) % NN;
  int b = idx / (NN * HH);
  const _Float16* mb = msgs + (size_t)b * EE * HH + h;
  float s = 0.0f;
  for (int snd = 0; snd < NN; ++snd) {
    if (snd == n) continue;
    int e = snd * (NN - 1) + n - (n > snd ? 1 : 0);
    s += (float)mb[(size_t)e * HH];
  }
  agg[idx] = s * (1.0f / 99.0f);
}

// ---------------------------------------------------------------- K4: GRU-style update
__global__ __launch_bounds__(256) void gru_kernel(
    const float* __restrict__ inputs, const float* __restrict__ hidden,
    const float* __restrict__ agg,
    const float* __restrict__ Wr, const float* __restrict__ Wi, const float* __restrict__ Wh,
    const float* __restrict__ irw, const float* __restrict__ irb,
    const float* __restrict__ iiw, const float* __restrict__ iib,
    const float* __restrict__ inw, const float* __restrict__ inb,
    float* __restrict__ newh) {
  int idx = blockIdx.x * blockDim.x + threadIdx.x;
  if (idx >= BB * NN * HH) return;
  int h  = idx & 63;
  int bn = idx >> 6;
  const float* x = inputs + bn * INSZ;
  const float* a = agg + bn * HH;
  float sr = irb[h], si = iib[h], sn = inb[h];
#pragma unroll
  for (int k = 0; k < INSZ; ++k) {
    float xv = x[k];
    sr += xv * irw[k * HH + h];
    si += xv * iiw[k * HH + h];
    sn += xv * inw[k * HH + h];
  }
  float ar = 0.f, ai = 0.f, ah = 0.f;
  for (int k = 0; k < HH; ++k) {
    float av = a[k];
    ar += av * Wr[k * HH + h];
    ai += av * Wi[k * HH + h];
    ah += av * Wh[k * HH + h];
  }
  float r   = 1.0f / (1.0f + expf(-(sr + ar)));
  float ig  = 1.0f / (1.0f + expf(-(si + ai)));
  float nn_ = tanhf(sn + r * ah);
  newh[idx] = (1.0f - ig) * nn_ + ig * hidden[idx];
}

// ---------------------------------------------------------------- K5: 64x64 matvec (+opt ReLU)
__global__ __launch_bounds__(256) void mv_kernel(const float* __restrict__ in,
                                                 const float* __restrict__ w,
                                                 const float* __restrict__ bias,
                                                 float* __restrict__ out, int relu) {
  int idx = blockIdx.x * blockDim.x + threadIdx.x;
  if (idx >= BB * NN * HH) return;
  int h  = idx & 63;
  int bn = idx >> 6;
  const float* v = in + bn * HH;
  float s = bias[h];
  for (int k = 0; k < HH; ++k) s += v[k] * w[k * HH + h];
  out[idx] = relu ? fmaxf(s, 0.0f) : s;
}

// ---------------------------------------------------------------- K6: final projection [B,6400]@[6400,200]
__global__ __launch_bounds__(256) void proj_kernel(const float* __restrict__ pred,
                                                   const float* __restrict__ pw,
                                                   const float* __restrict__ pb,
                                                   float* __restrict__ distr) {
  int b = blockIdx.x;
  int j = threadIdx.x;
  if (j >= 2 * NN) return;
  const float* p = pred + (size_t)b * NN * HH;
  float s = pb[j];
  for (int i = 0; i < NN * HH; ++i) s += p[i] * pw[(size_t)i * (2 * NN) + j];
  distr[b * 2 * NN + j] = s;
}

// ---------------------------------------------------------------- launch
extern "C" void kernel_launch(void* const* d_in, const int* in_sizes, int n_in,
                              void* d_out, int out_size, void* d_ws, size_t ws_size,
                              hipStream_t stream) {
  (void)in_sizes; (void)n_in; (void)out_size; (void)ws_size;
  const float* inputs = (const float*)d_in[0];
  const float* hidden = (const float*)d_in[1];
  const float* elog   = (const float*)d_in[2];
  const float* fc1w   = (const float*)d_in[3];
  const float* fc1b   = (const float*)d_in[4];
  const float* fc2w   = (const float*)d_in[5];
  const float* fc2b   = (const float*)d_in[6];
  const float* Wr     = (const float*)d_in[7];
  const float* Wi     = (const float*)d_in[8];
  const float* Wh     = (const float*)d_in[9];
  const float* irw = (const float*)d_in[10]; const float* irb = (const float*)d_in[11];
  const float* iiw = (const float*)d_in[12]; const float* iib = (const float*)d_in[13];
  const float* inw = (const float*)d_in[14]; const float* inb = (const float*)d_in[15];
  const float* m1w = (const float*)d_in[16]; const float* m1b = (const float*)d_in[17];
  const float* m2w = (const float*)d_in[18]; const float* m2b = (const float*)d_in[19];
  const float* pw  = (const float*)d_in[20]; const float* pb  = (const float*)d_in[21];
  const int* send = (const int*)d_in[22];
  const int* recv = (const int*)d_in[23];

  float* out   = (float*)d_out;
  float* distr = out;               // [B, 2N]
  float* newh  = out + BB * 2 * NN; // [B, N, H]

  char* ws = (char*)d_ws;
  size_t off = 0;
  _Float16* h16  = (_Float16*)(ws + off); off += (size_t)BB * NN * HH * 2;  // 409,600 B
  float*    gate = (float*)(ws + off);    off += (size_t)BB * EE * 4;       // 1,267,200 B
  _Float16* msgs = (_Float16*)(ws + off); off += (size_t)BB * EE * HH * 2;  // 40,550,400 B
  float*    agg  = (float*)(ws + off);    off += (size_t)BB * NN * HH * 4;
  float*    tmp  = (float*)(ws + off);    off += (size_t)BB * NN * HH * 4;
  float*    pred = (float*)(ws + off);

  const int nBNH = BB * NN * HH;
  gate_kernel<<<(BB * EE + 255) / 256, 256, 0, stream>>>(elog, gate);
  h2f16_kernel<<<(nBNH + 255) / 256, 256, 0, stream>>>(hidden, h16, nBNH);
  edge_mlp_kernel<<<BB * BLKS_PB, 256, 0, stream>>>(h16, gate, fc1w, fc1b, fc2w, fc2b,
                                                    send, recv, msgs);
  agg_kernel<<<(nBNH + 255) / 256, 256, 0, stream>>>(msgs, agg);
  gru_kernel<<<(nBNH + 255) / 256, 256, 0, stream>>>(inputs, hidden, agg, Wr, Wi, Wh,
                                                     irw, irb, iiw, iib, inw, inb, newh);
  mv_kernel<<<(nBNH + 255) / 256, 256, 0, stream>>>(newh, m1w, m1b, tmp, 1);
  mv_kernel<<<(nBNH + 255) / 256, 256, 0, stream>>>(tmp, m2w, m2b, pred, 0);
  proj_kernel<<<BB, 256, 0, stream>>>(pred, pw, pb, distr);
}
